// AdGraphMultiAttn_42984032698883
// MI455X (gfx1250) — matrix operand order repaired and checked by reading, not compiled, not run on hardware
//
#include <hip/hip_runtime.h>
#include <hip/hip_bf16.h>

#define NN 10000
#define NE 320000
#define LD 256
#define NH 4

typedef __attribute__((ext_vector_type(16))) _Float16 v16h;
typedef __attribute__((ext_vector_type(8)))  float    v8f;
typedef __attribute__((ext_vector_type(4)))  unsigned v4u;
typedef __attribute__((ext_vector_type(8)))  int      v8i;
typedef __attribute__((ext_vector_type(4)))  int      v4i;

// ---------- device helpers ----------
__device__ __forceinline__ float gelu_f(float x){
  float x3 = x*x*x;
  return 0.5f*x*(1.0f + tanhf(0.7978845608028654f*(x + 0.044715f*x3)));
}
__device__ __forceinline__ float sigm(float x){ return 1.0f/(1.0f+__expf(-x)); }
__device__ __forceinline__ void atomicMaxF(float* a, float v){
  if (v >= 0.f) atomicMax((int*)a, __float_as_int(v));
  else          atomicMin((unsigned int*)a, (unsigned int)__float_as_int(v));
}

// ---------- CDNA5 Tensor Data Mover: 2D tile global->LDS ----------
// Moves `rows` rows of 512B (64 x 8B units), row stride 512B, to LDS. Wave-level op,
// tracked by TENSORcnt. Descriptor per CDNA5 ISA.
__device__ __forceinline__ void tdm_raw(v4u g0, v8i g1){
#if __clang_major__ >= 23
  __builtin_amdgcn_tensor_load_to_lds(g0, g1, (v4i)0, (v4i)0, (v8i)0, 0);
#else
  __builtin_amdgcn_tensor_load_to_lds(g0, g1, (v4i)0, (v4i)0, 0);
#endif
}
__device__ __forceinline__ void tdm_tile_2d(unsigned lds_off, const void* g, unsigned rows){
  unsigned long long ga = (unsigned long long)(size_t)g;
  v4u g0 = { 1u,                                     // count=1, user mode
             lds_off,                                // LDS byte address
             (unsigned)ga,                           // global_addr[31:0]
             ((unsigned)(ga >> 32) & 0x01FFFFFFu) | 0x80000000u }; // addr[56:32] | type=2
  v8i g1 = { (int)(3u << 16),        // data_size=8B
             (int)(64u << 16),       // tensor_dim0 = 64 units (512B row)
             (int)(rows << 16),      // tensor_dim1 = rows
             (int)(64u << 16),       // tile_dim0 = 64 units
             (int)rows,              // tile_dim1 = rows (tile_dim2 = 0)
             64,                     // tensor_dim0_stride = 64 units (512B)
             0, 0 };
  tdm_raw(g0, g1);
}

// ---------- f16 conversion kernels ----------
__global__ void cvt_x_f16(const float* __restrict__ x, _Float16* __restrict__ xh, int n){
  int i = blockIdx.x*256 + threadIdx.x;
  if (i < n) xh[i] = (_Float16)x[i];
}
// W fp32 [k][n] row-major -> Wt f16 [n][k] (rows contiguous in k)
__global__ void cvt_wT_f16(const float* __restrict__ W, _Float16* __restrict__ Wt){
  int n = blockIdx.x, k = threadIdx.x;
  Wt[n*256 + k] = (_Float16)W[k*256 + n];
}

// ---------- batched WMMA GEMM with TDM double-buffered B tiles ----------
// out[j][M,256] = A[M,256] @ B[j]^T + bias[j]; A/B f16, B pre-transposed [n][k].
// block = 128 threads (4 wave32). Tile: 16 rows x 64 cols; A resident across batch.
struct GB { const _Float16* B[4]; const float* bias[4]; float* out[4]; int n; };

__global__ void __launch_bounds__(128)
gemm256_wmma_tdm(const _Float16* __restrict__ A, GB gb, int M){
  __shared__ _Float16 Al[16*256];      //  8 KB [m][k]
  __shared__ _Float16 Bl[2][64*256];   // 2 x 32 KB [n][k], double-buffered
  const int tid  = threadIdx.x;
  const int wave = tid >> 5;
  const int lane = tid & 31;
  const int m0 = blockIdx.x * 16;
  const int c0 = blockIdx.y * 64;

  if (tid < 32){                        // wave 0 drives the TDM
    tdm_tile_2d((unsigned)(size_t)Al,        (const char*)A       + (size_t)m0*512, 16);
    tdm_tile_2d((unsigned)(size_t)&Bl[0][0], (const char*)gb.B[0] + (size_t)c0*512, 64);
  }

  const int m  = lane & 15;
  const int hi = lane >> 4;
  const int nl = wave*16 + (lane & 15);
  const int cg = c0 + wave*16 + (lane & 15);

  #pragma unroll
  for (int j = 0; j < 4; ++j){
    if (j >= gb.n) break;
    if (tid < 32){
      if (j + 1 < gb.n){                // prefetch next B tile into alternate buffer
        tdm_tile_2d((unsigned)(size_t)&Bl[(j+1)&1][0],
                    (const char*)gb.B[j+1] + (size_t)c0*512, 64);
        __builtin_amdgcn_s_wait_tensorcnt(1);   // current tile (and A) landed
      } else {
        __builtin_amdgcn_s_wait_tensorcnt(0);
      }
    }
    __syncthreads();

    const _Float16* Bc = &Bl[j & 1][0];
    v8f acc = {0.f,0.f,0.f,0.f,0.f,0.f,0.f,0.f};
    #pragma unroll
    for (int t = 0; t < 8; ++t){                    // K = 8 x 32
      const int kb = t*32;
      v16h af, bf;
      #pragma unroll
      for (int r = 0; r < 8; ++r){
        // A 16x32 f16: lanes0-15 K in {0-7,16-23}, lanes16-31 K in {8-15,24-31}
        int ka = kb + ((r & 4) ? 16 : 0) + hi*8 + 2*(r & 3);
        af[2*r]   = Al[m*256 + ka];
        af[2*r+1] = Al[m*256 + ka + 1];
        // B 32x16 f16: lanes0-15 K in {0-7,16-23}, lanes16-31 K in {8-15,24-31}
        int kn = kb + ((r & 4) ? 16 : 0) + hi*8 + 2*(r & 3);
        bf[2*r]   = Bc[nl*256 + kn];
        bf[2*r+1] = Bc[nl*256 + kn + 1];
      }
      acc = __builtin_amdgcn_wmma_f32_16x16x32_f16(false, af, false, bf,
                                                   (short)0, acc, false, false);
    }
    const float bv = gb.bias[j] ? gb.bias[j][cg] : 0.0f;
    float* Co = gb.out[j];
    #pragma unroll
    for (int r = 0; r < 8; ++r){                    // D: M = r + 8*hi, N = lane&15
      int rg = m0 + r + 8*hi;
      if (rg < M) Co[(size_t)rg*256 + cg] = acc[r] + bv;
    }
    __syncthreads();                                // buffer reuse protection
  }
}

// ---------- edge attention ----------
__global__ void init_seg(float* mmax, float* ssum, float* deg){
  int i = blockIdx.x*256 + threadIdx.x;
  if (i < NN*NH){ mmax[i] = -__builtin_inff(); ssum[i] = 0.f; }
  if (i < NN) deg[i] = 0.f;
}

__global__ void edge_score(const float* __restrict__ q, const float* __restrict__ k,
                           const int* __restrict__ src, const int* __restrict__ dst,
                           const int* __restrict__ gate,
                           float* __restrict__ score, float* __restrict__ mmax){
  int idx = blockIdx.x*256 + threadIdx.x;
  if (idx >= NE*NH) return;
  int e = idx >> 2, h = idx & 3;
  if (gate && !gate[e]) { score[idx] = -__builtin_inff(); return; }
  int d = dst[e], s = src[e];
  const float* qr = q + (size_t)d*256 + h*64;
  const float* kr = k + (size_t)s*256 + h*64;
  float acc = 0.f;
  #pragma unroll 8
  for (int c = 0; c < 64; ++c) acc += qr[c]*kr[c];
  acc *= 0.125f;
  score[idx] = acc;
  atomicMaxF(&mmax[d*NH + h], acc);
}

__global__ void edge_sumexp(float* __restrict__ score, const int* __restrict__ dst,
                            const int* __restrict__ gate,
                            const float* __restrict__ mmax, float* __restrict__ ssum){
  int idx = blockIdx.x*256 + threadIdx.x;
  if (idx >= NE*NH) return;
  int e = idx >> 2, h = idx & 3;
  if (gate && !gate[e]) return;
  int d = dst[e];
  float m = mmax[d*NH + h];
  if (m < -1e30f) m = 0.f;
  float ex = __expf(score[idx] - m);
  score[idx] = ex;
  atomicAdd(&ssum[d*NH + h], ex);
}

__global__ void edge_agg(const float* __restrict__ score, const float* __restrict__ ssum,
                         const float* __restrict__ v,
                         const int* __restrict__ src, const int* __restrict__ dst,
                         const int* __restrict__ gate, float* __restrict__ hout){
  int e = blockIdx.x;
  if (gate && !gate[e]) return;
  int c = threadIdx.x, h = c >> 6;
  int d = dst[e], s = src[e];
  float den = ssum[d*NH + h]; den = den > 0.f ? den : 1.f;
  float alpha = score[(size_t)e*NH + h] / den;
  atomicAdd(&hout[(size_t)d*256 + c], alpha * v[(size_t)s*256 + c]);
}

__global__ void edge_meta(const float* __restrict__ score, const float* __restrict__ ssum,
                          const int* __restrict__ dst,
                          float* __restrict__ amean, float* __restrict__ deg){
  int e = blockIdx.x*256 + threadIdx.x;
  if (e >= NE) return;
  int d = dst[e];
  float am = 0.f;
  #pragma unroll
  for (int h = 0; h < NH; ++h){
    float den = ssum[d*NH + h]; den = den > 0.f ? den : 1.f;
    am += score[(size_t)e*NH + h] / den;
  }
  amean[e] = am * 0.25f;
  atomicAdd(&deg[d], 1.f);
}

__global__ void gate_k(const float* __restrict__ amean, const float* __restrict__ deg,
                       const int* __restrict__ dst, const float* __restrict__ theta,
                       int* __restrict__ gate){
  int e = blockIdx.x*256 + threadIdx.x;
  if (e >= NE) return;
  float bm = 0.25f*(sigm(theta[0])+sigm(theta[1])+sigm(theta[2])+sigm(theta[3]));
  float thr = bm / fmaxf(deg[dst[e]], 1.f);
  gate[e] = (amean[e] >= thr) ? 1 : 0;
}

// ---------- LayerNorm / GELU / residual ----------
__global__ void ln_gelu_res(const float* __restrict__ hin, const float* __restrict__ g,
                            const float* __restrict__ b, float* __restrict__ xcur,
                            int doLN, int doRes, int twoGelu){
  int n = blockIdx.x, t = threadIdx.x;
  float val = hin[(size_t)n*256 + t];
  __shared__ float red[256];
  if (doLN){
    red[t] = val; __syncthreads();
    for (int s = 128; s > 0; s >>= 1){ if (t < s) red[t] += red[t+s]; __syncthreads(); }
    float mu = red[0]*(1.f/256.f); __syncthreads();
    float dv = val - mu;
    red[t] = dv*dv; __syncthreads();
    for (int s = 128; s > 0; s >>= 1){ if (t < s) red[t] += red[t+s]; __syncthreads(); }
    float var = red[0]*(1.f/256.f);
    val = dv*rsqrtf(var + 1e-5f)*g[t] + b[t];
  }
  val = gelu_f(val);
  if (twoGelu) val = gelu_f(val);
  if (doRes) val += xcur[(size_t)n*256 + t];
  xcur[(size_t)n*256 + t] = val;
}

// ---------- pooling ----------
__global__ void pool_gate_ew(float* __restrict__ xV, const float* __restrict__ xU, int n){
  int i = blockIdx.x*256 + threadIdx.x;
  if (i < n) xV[i] = tanhf(xV[i]) * sigm(xU[i]);
}

__global__ void pool_logits(const float* __restrict__ a, const float* __restrict__ W,
                            float* __restrict__ logits){
  int idx = blockIdx.x*256 + threadIdx.x;
  if (idx >= NN*2) return;
  int n = idx >> 1, cls = idx & 1;
  float s = 0.f;
  for (int k = 0; k < 256; ++k) s += a[(size_t)n*256 + k] * W[k*2 + cls];
  logits[idx] = s;
}

__global__ void pool_softmax(const float* __restrict__ logits, float* __restrict__ attn){
  int cls = blockIdx.x, t = threadIdx.x;
  __shared__ float red[256];
  __shared__ float m_s, s_s;
  float m = -1e30f;
  for (int i = t; i < NN; i += 256) m = fmaxf(m, logits[i*2 + cls]);
  red[t] = m; __syncthreads();
  for (int s = 128; s > 0; s >>= 1){ if (t < s) red[t] = fmaxf(red[t], red[t+s]); __syncthreads(); }
  if (t == 0) m_s = red[0]; __syncthreads();
  float sum = 0.f;
  for (int i = t; i < NN; i += 256) sum += __expf(logits[i*2 + cls] - m_s);
  red[t] = sum; __syncthreads();
  for (int s = 128; s > 0; s >>= 1){ if (t < s) red[t] += red[t+s]; __syncthreads(); }
  if (t == 0) s_s = red[0]; __syncthreads();
  for (int i = t; i < NN; i += 256) attn[i*2 + cls] = __expf(logits[i*2 + cls] - m_s) / s_s;
}

__global__ void pool_M(const float* __restrict__ attn, const float* __restrict__ x,
                       float* __restrict__ Mb){
  int cls = blockIdx.x, c = threadIdx.x;
  float s = 0.f;
  for (int n = 0; n < NN; ++n) s += attn[n*2 + cls] * x[(size_t)n*256 + c];
  Mb[cls*256 + c] = s;
}

__global__ void pool_y(const float* __restrict__ Mb, const float* __restrict__ Wc,
                       const float* __restrict__ bc, float* __restrict__ y){
  int cls = blockIdx.x, c = threadIdx.x;
  __shared__ float red[256];
  red[c] = Mb[cls*256 + c] * Wc[cls*256 + c]; __syncthreads();
  for (int s = 128; s > 0; s >>= 1){ if (c < s) red[c] += red[c+s]; __syncthreads(); }
  if (c == 0) y[cls] = red[0] + bc[cls];
}

// ---------- host side ----------
static inline int cdiv(int a, int b){ return (a + b - 1) / b; }

extern "C" void kernel_launch(void* const* d_in, const int* in_sizes, int n_in,
                              void* d_out, int out_size, void* d_ws, size_t ws_size,
                              hipStream_t stream) {
  const float* X   = (const float*)d_in[0];
  const int*   EI  = (const int*)d_in[1];
  const int* src = EI;
  const int* dst = EI + NE;

  auto F = [&](int i){ return (const float*)d_in[i]; };
  // adapt (3..13): Wk,Wq,Ws,Wv,bk,bq,bs,bv,ln_b,ln_g,theta
  const float *abq=F(8),*abk=F(7),*abv=F(10),*abs_=F(9);
  const float *aln_b=F(11),*aln_g=F(12),*atheta=F(13);
  // out (44..51): Wk,Wq,Ws,bk,Wv,bq,bs,bv
  const float *obq=F(49),*obk=F(47),*obv=F(51),*obs=F(50);
  // pool (52..56): U,V,W,Wc,bc
  const float *pU=F(52),*pV=F(53),*pW=F(54),*pWc=F(55),*pbc=F(56);

  // workspace carve (float elements)
  float* ws = (float*)d_ws;
  float* xcur  = ws;                      // 2,560,000
  float* qb    = ws +  2560000;
  float* kb    = ws +  5120000;
  float* vb    = ws +  7680000;
  float* hb    = ws + 10240000;
  float* score = ws + 12800000;           // E*H (reused as logits)
  float* mmax  = ws + 14080000;           // N*H
  float* ssum  = ws + 14120000;
  float* deg   = ws + 14160000;
  float* amean = ws + 14170000;           // E
  int*   gate  = (int*)(ws + 14490000);   // E ints
  float* Mbuf  = ws + 14810000;           // 512
  _Float16* xh = (_Float16*)(ws + 14811000);  // 2,560,000 halves
  _Float16* wh = (_Float16*)(ws + 16091000);  // 22 x 65,536 halves

  float* y_out    = (float*)d_out;
  float* attn_out = y_out + 2;
  float* x_out    = attn_out + NN*2;

  // --- one-time weight convert+transpose to f16 ---
  // order: adapt Wq,Wk,Wv,Ws | layer{0,1,2} Wq,Wk,Wv,Ws | out Wq,Wk,Wv,Ws | pool V,U
  const float* wsrc[22];
  wsrc[0]=F(4); wsrc[1]=F(3); wsrc[2]=F(6); wsrc[3]=F(5);          // adapt
  for (int l = 0; l < 3; ++l){
    int b = 14 + 10*l;                                             // Wk,Wq,Ws,Wv,...
    wsrc[4+4*l+0]=F(b+1); wsrc[4+4*l+1]=F(b+0);
    wsrc[4+4*l+2]=F(b+3); wsrc[4+4*l+3]=F(b+2);
  }
  wsrc[16]=F(45); wsrc[17]=F(44); wsrc[18]=F(48); wsrc[19]=F(46);  // out
  wsrc[20]=pV; wsrc[21]=pU;                                        // pool
  for (int i = 0; i < 22; ++i)
    cvt_wT_f16<<<256, 256, 0, stream>>>(wsrc[i], wh + (size_t)i*65536);
  auto WJ = [&](int j){ return (const _Float16*)(wh + (size_t)j*65536); };

  hipMemcpyAsync(xcur, X, (size_t)NN*256*sizeof(float), hipMemcpyDeviceToDevice, stream);

  dim3 gg(NN/16, 4);
  auto run_conv = [&](int widx,
                      const float* bq,const float* bk,const float* bv,const float* bs,
                      const int* gmask){
    cvt_x_f16<<<cdiv(NN*256,256), 256, 0, stream>>>(xcur, xh, NN*256);
    GB gb;
    gb.B[0]=WJ(widx); gb.B[1]=WJ(widx+1); gb.B[2]=WJ(widx+2); gb.B[3]=WJ(widx+3);
    gb.bias[0]=bq; gb.bias[1]=bk; gb.bias[2]=bv; gb.bias[3]=bs;
    gb.out[0]=qb; gb.out[1]=kb; gb.out[2]=vb; gb.out[3]=hb;
    gb.n = 4;
    gemm256_wmma_tdm<<<gg, 128, 0, stream>>>(xh, gb, NN);
    init_seg   <<<cdiv(NN*NH,256), 256, 0, stream>>>(mmax, ssum, deg);
    edge_score <<<cdiv(NE*NH,256), 256, 0, stream>>>(qb, kb, src, dst, gmask, score, mmax);
    edge_sumexp<<<cdiv(NE*NH,256), 256, 0, stream>>>(score, dst, gmask, mmax, ssum);
    edge_agg   <<<NE, 256, 0, stream>>>(score, ssum, vb, src, dst, gmask, hb);
  };

  // --- AdaptFullConv (no mask) ---
  run_conv(0, abq, abk, abv, abs_, nullptr);
  edge_meta<<<cdiv(NE,256), 256, 0, stream>>>(score, ssum, dst, amean, deg);
  gate_k   <<<cdiv(NE,256), 256, 0, stream>>>(amean, deg, dst, atheta, gate);
  ln_gelu_res<<<NN, 256, 0, stream>>>(hb, aln_g, aln_b, xcur, 1, 1, 0);

  // --- 3 residual layers on gated edges ---
  for (int l = 0; l < 3; ++l){
    int b = 14 + 10*l;
    run_conv(4 + 4*l, F(b+5), F(b+4), F(b+7), F(b+6), gate);
    ln_gelu_res<<<NN, 256, 0, stream>>>(hb, F(b+9), F(b+8), xcur, 1, 1, 0);
  }

  // --- out conv (gated) + double GELU ---
  run_conv(16, obq, obk, obv, obs, gate);
  ln_gelu_res<<<NN, 256, 0, stream>>>(hb, nullptr, nullptr, xcur, 0, 0, 1);

  // --- pooling ---
  cvt_x_f16<<<cdiv(NN*256,256), 256, 0, stream>>>(xcur, xh, NN*256);
  GB pg;
  pg.B[0]=WJ(20); pg.B[1]=WJ(21); pg.B[2]=nullptr; pg.B[3]=nullptr;
  pg.bias[0]=nullptr; pg.bias[1]=nullptr; pg.bias[2]=nullptr; pg.bias[3]=nullptr;
  pg.out[0]=qb; pg.out[1]=kb; pg.out[2]=nullptr; pg.out[3]=nullptr;
  pg.n = 2;
  gemm256_wmma_tdm<<<gg, 128, 0, stream>>>(xh, pg, NN);    // x@V, x@U
  pool_gate_ew<<<cdiv(NN*256,256), 256, 0, stream>>>(qb, kb, NN*256);
  pool_logits <<<cdiv(NN*2,256), 256, 0, stream>>>(qb, pW, score);
  pool_softmax<<<2, 256, 0, stream>>>(score, attn_out);
  pool_M      <<<2, 256, 0, stream>>>(attn_out, xcur, Mbuf);
  pool_y      <<<2, 256, 0, stream>>>(Mbuf, pWc, pbc, y_out);

  hipMemcpyAsync(x_out, xcur, (size_t)NN*256*sizeof(float), hipMemcpyDeviceToDevice, stream);
}
